// Transformer_8804682956970
// MI455X (gfx1250) — matrix operand order repaired and checked by reading, not compiled, not run on hardware
//
#include <hip/hip_runtime.h>

// ---------------------------------------------------------------------------
// Types for CDNA5 WMMA (wave32): v_wmma_f32_16x16x32_bf16
// ---------------------------------------------------------------------------
typedef __bf16 bf16_t;
typedef bf16_t v16bf __attribute__((ext_vector_type(16)));
typedef bf16_t bf16x8 __attribute__((ext_vector_type(8)));
typedef float v8f __attribute__((ext_vector_type(8)));

__device__ __forceinline__ bf16_t f2bf(float f) {
  union { float f; unsigned u; } cvt; cvt.f = f;
  unsigned u = cvt.u;
  u = (u + 0x7FFFu + ((u >> 16) & 1u)) >> 16;  // round-to-nearest-even
  union { unsigned short s; bf16_t b; } o; o.s = (unsigned short)u;
  return o.b;
}

// Async DMA: global -> LDS, 16 bytes per active lane, tracked by ASYNCcnt.
// (inline asm: portable across ROCm 7.2 / amdgpu-toolchain per CDNA5_HIP.md)
__device__ __forceinline__ void async_g2l_b128(void* ldsPtr, const void* gPtr) {
  unsigned lds = (unsigned)(unsigned long long)ldsPtr;  // low 32b = LDS offset
  unsigned long long ga = (unsigned long long)gPtr;
  asm volatile("global_load_async_to_lds_b128 %0, %1, off"
               :: "v"(lds), "v"(ga)
               : "memory");
}
__device__ __forceinline__ void wait_async3() {
#if __has_builtin(__builtin_amdgcn_s_wait_asynccnt)
  __builtin_amdgcn_s_wait_asynccnt(3);
#else
  asm volatile("s_wait_asynccnt 0x3" ::: "memory");
#endif
}
__device__ __forceinline__ void wait_async0() {
#if __has_builtin(__builtin_amdgcn_s_wait_asynccnt)
  __builtin_amdgcn_s_wait_asynccnt(0);
#else
  asm volatile("s_wait_asynccnt 0x0" ::: "memory");
#endif
}

// ---------------------------------------------------------------------------
// LDS-staged, async double-buffered GEMM (B given as W[N][K] row-major):
//   C[m,n] = alpha * sum_k A[m,k] * W[n,k] (+ bias[n])
// Requirements (guaranteed by all call sites): M%64==0, N%128==0, K%32==0.
// Block = 256 threads = 8 waves (2x4); block tile 64x128; wave tile 32x32:
// 2 A-fragments x 2 B-fragments -> 4 WMMAs per 32-k step (8 ds_load_b128),
// 2:1 LDS:matrix-op ratio. No divergence: EXEC all-1s at every WMMA, no
// early exits (barriers safe).
// Per stage: A tile 64x32 (4KB, 256 x16B chunks = 1/thread) + B tile 128x32
// (8KB, 512 chunks = 2/thread), double buffered (24KB LDS). Each wave issues
// exactly 3 async-b128 ops per stage -> steady-state s_wait_asynccnt 3.
// ---------------------------------------------------------------------------
__global__ void __launch_bounds__(256) gemm_bf16_lds(
    const bf16_t* __restrict__ A, const bf16_t* __restrict__ Bm,
    const float* __restrict__ bias, float* __restrict__ Cf,
    bf16_t* __restrict__ Cbf, int M, int N, int K, int lda, int ldb, int ldc,
    long long aStrideZ, long long bStrideZ, long long cStrideB,
    long long cStrideH, int headsPerZ, float alpha, int doRelu) {
  __shared__ bf16_t As[2][64][32];
  __shared__ bf16_t Bs[2][128][32];

  const int z = blockIdx.z;
  const bf16_t* Az = A + (long long)z * aStrideZ;
  const bf16_t* Bz = Bm + (long long)z * bStrideZ;
  const long long cOff = (long long)(z / headsPerZ) * cStrideB +
                         (long long)(z % headsPerZ) * cStrideH;

  const int tid = threadIdx.x;
  const int wave = tid >> 5;
  const int lane = tid & 31;
  const int lo = lane & 15;
  const int hi = lane >> 4;
  const int wr = wave >> 2;        // wave row group (0..1)
  const int wc = wave & 3;         // wave col group (0..3)
  const int rowBase = blockIdx.y * 64;
  const int colBase = blockIdx.x * 128;

  // Per-stage async issue: 1 A chunk + 2 B chunks per thread (16B each).
  const int aRow = tid >> 2, aSeg = tid & 3;                 // A: 256 chunks
  const int bCol0 = tid >> 2, bSeg0 = tid & 3;               // B: chunks tid
  const int bCol1 = (tid + 256) >> 2, bSeg1 = (tid + 256) & 3;  // and tid+256

  const int nk = K >> 5;
  auto issue = [&](int kt) {
    const int buf = kt & 1;
    const int k0 = kt << 5;
    async_g2l_b128(&As[buf][aRow][aSeg * 8],
                   Az + (long long)(rowBase + aRow) * lda + k0 + aSeg * 8);
    async_g2l_b128(&Bs[buf][bCol0][bSeg0 * 8],
                   Bz + (long long)(colBase + bCol0) * ldb + k0 + bSeg0 * 8);
    async_g2l_b128(&Bs[buf][bCol1][bSeg1 * 8],
                   Bz + (long long)(colBase + bCol1) * ldb + k0 + bSeg1 * 8);
  };

  v8f c00 = {}, c01 = {}, c10 = {}, c11 = {};
  const int fRow = wr * 32 + lo;   // A fragment rows: fRow and fRow+16
  const int fCol = wc * 32 + lo;   // B fragment cols: fCol and fCol+16

  issue(0);
  for (int kt = 0; kt < nk; ++kt) {
    const int buf = kt & 1;
    if (kt + 1 < nk) {             // prefetch next stage into other buffer
      issue(kt + 1);
      wait_async3();               // my current-stage DMAs have landed
    } else {
      wait_async0();
    }
    __syncthreads();               // everyone's current-stage tile is in LDS

    // A fragments (16x32 bf16): elems 0..7 -> K=hi*16+j ; 8..15 -> hi*16+8+j
    bf16x8 a0l = *(const bf16x8*)&As[buf][fRow][hi * 16];
    bf16x8 a0h = *(const bf16x8*)&As[buf][fRow][hi * 16 + 8];
    bf16x8 a1l = *(const bf16x8*)&As[buf][fRow + 16][hi * 16];
    bf16x8 a1h = *(const bf16x8*)&As[buf][fRow + 16][hi * 16 + 8];
    v16bf a0, a1;
#pragma unroll
    for (int j = 0; j < 8; ++j) {
      a0[j] = a0l[j]; a0[j + 8] = a0h[j];
      a1[j] = a1l[j]; a1[j + 8] = a1h[j];
    }
    // B fragments (32x16 bf16): elem j -> K = hi*16 + j
    v16bf b0 = *(const v16bf*)&Bs[buf][fCol][hi * 16];
    v16bf b1 = *(const v16bf*)&Bs[buf][fCol + 16][hi * 16];

    c00 = __builtin_amdgcn_wmma_f32_16x16x32_bf16(false, a0, false, b0,
                                                  (short)0, c00, false, false);
    c01 = __builtin_amdgcn_wmma_f32_16x16x32_bf16(false, a0, false, b1,
                                                  (short)0, c01, false, false);
    c10 = __builtin_amdgcn_wmma_f32_16x16x32_bf16(false, a1, false, b0,
                                                  (short)0, c10, false, false);
    c11 = __builtin_amdgcn_wmma_f32_16x16x32_bf16(false, a1, false, b1,
                                                  (short)0, c11, false, false);
    __syncthreads();               // tile fully consumed; buf reusable
  }

  // C layout: VGPR r -> row = rowFrag + hi*8 + r, col = colFrag + lo
  const int colA = colBase + wc * 32 + lo;
  const float biasA = bias ? bias[colA] : 0.f;
  const float biasB = bias ? bias[colA + 16] : 0.f;
  const int rowA = rowBase + wr * 32 + hi * 8;
#pragma unroll
  for (int r = 0; r < 8; ++r) {
    const long long i0 = cOff + (long long)(rowA + r) * ldc + colA;
    const long long i1 = cOff + (long long)(rowA + 16 + r) * ldc + colA;
    float v00 = c00[r] * alpha + biasA;
    float v01 = c01[r] * alpha + biasB;
    float v10 = c10[r] * alpha + biasA;
    float v11 = c11[r] * alpha + biasB;
    if (doRelu) {
      v00 = v00 > 0.f ? v00 : 0.f; v01 = v01 > 0.f ? v01 : 0.f;
      v10 = v10 > 0.f ? v10 : 0.f; v11 = v11 > 0.f ? v11 : 0.f;
    }
    if (Cf) {
      Cf[i0] = v00; Cf[i0 + 16] = v01;
      Cf[i1] = v10; Cf[i1 + 16] = v11;
    }
    if (Cbf) {
      Cbf[i0] = f2bf(v00); Cbf[i0 + 16] = f2bf(v01);
      Cbf[i1] = f2bf(v10); Cbf[i1 + 16] = f2bf(v11);
    }
  }
}

// ---------------------------------------------------------------------------
// Direct-load GEMM with B given as B[K][N] row-major (only used for the small
// P*V GEMM: M=512, N=64, K=512 per (b,h)). Guarded tiles for N=64.
// ---------------------------------------------------------------------------
__global__ void __launch_bounds__(256) gemm_bf16_bn(
    const bf16_t* __restrict__ A, const bf16_t* __restrict__ Bm,
    const float* __restrict__ bias, float* __restrict__ Cf,
    bf16_t* __restrict__ Cbf, int M, int N, int K, int lda, int ldb, int ldc,
    long long aStrideZ, long long bStrideZ, long long cStrideB,
    long long cStrideH, int headsPerZ, float alpha, int doRelu) {
  const int z = blockIdx.z;
  const bf16_t* Az = A + (long long)z * aStrideZ;
  const bf16_t* Bz = Bm + (long long)z * bStrideZ;
  const long long cOff = (long long)(z / headsPerZ) * cStrideB +
                         (long long)(z % headsPerZ) * cStrideH;

  const int wave = threadIdx.x >> 5;
  const int lane = threadIdx.x & 31;
  const int lo = lane & 15;
  const int hi = lane >> 4;
  const int row0 = blockIdx.y * 32 + (wave >> 2) * 16;
  const int col0 = blockIdx.x * 128 + (wave & 3) * 32;
  if (row0 >= M || col0 >= N) return;  // wave-uniform
  const bool do1 = (col0 + 16) < N;

  v8f c0 = {};
  v8f c1 = {};
  const bf16_t* aPtr = Az + (long long)(row0 + lo) * lda + hi * 16;

  for (int k0 = 0; k0 < K; k0 += 32) {
    if (k0 + 32 < K) __builtin_prefetch(aPtr + k0 + 32, 0, 1);
    bf16x8 aLoV = *(const bf16x8*)(aPtr + k0);
    bf16x8 aHiV = *(const bf16x8*)(aPtr + k0 + 8);
    v16bf a;
#pragma unroll
    for (int j = 0; j < 8; ++j) { a[j] = aLoV[j]; a[j + 8] = aHiV[j]; }

    const int kk = k0 + hi * 16;
    v16bf b0;
#pragma unroll
    for (int j = 0; j < 16; ++j)
      b0[j] = Bz[(long long)(kk + j) * ldb + (col0 + lo)];
    c0 = __builtin_amdgcn_wmma_f32_16x16x32_bf16(false, a, false, b0, (short)0,
                                                 c0, false, false);
    if (do1) {
      v16bf b1;
#pragma unroll
      for (int j = 0; j < 16; ++j)
        b1[j] = Bz[(long long)(kk + j) * ldb + (col0 + 16 + lo)];
      c1 = __builtin_amdgcn_wmma_f32_16x16x32_bf16(false, a, false, b1,
                                                   (short)0, c1, false, false);
    }
  }

  const int colA = col0 + lo;
  const float biasA = bias ? bias[colA] : 0.f;
  const float biasB = (do1 && bias) ? bias[colA + 16] : 0.f;
#pragma unroll
  for (int r = 0; r < 8; ++r) {
    const int row = row0 + hi * 8 + r;
    const long long idx = cOff + (long long)row * ldc + colA;
    float v = c0[r] * alpha + biasA;
    if (doRelu) v = v > 0.f ? v : 0.f;
    if (Cf) Cf[idx] = v;
    if (Cbf) Cbf[idx] = f2bf(v);
    if (do1) {
      float w = c1[r] * alpha + biasB;
      if (doRelu) w = w > 0.f ? w : 0.f;
      if (Cf) Cf[idx + 16] = w;
      if (Cbf) Cbf[idx + 16] = f2bf(w);
    }
  }
}

// ---------------------------------------------------------------------------
// Row softmax over 512 columns with optional int mask (mask==0 -> -FLT_MAX).
// ---------------------------------------------------------------------------
__global__ void __launch_bounds__(256) softmax_mask(
    const float* __restrict__ Sc, const int* __restrict__ mask,
    bf16_t* __restrict__ P, int rowsPerZ, int headsPerBatch,
    long long maskBatchStride, long long maskRowStride) {
  const int row = blockIdx.x;
  const int t = threadIdx.x;
  const int q = row % rowsPerZ;
  const int z = row / rowsPerZ;
  const int b = z / headsPerBatch;
  const float* src = Sc + (long long)row * 512;
  float v0 = src[t];
  float v1 = src[t + 256];
  if (mask) {
    const int* mrow =
        mask + (long long)b * maskBatchStride + (long long)q * maskRowStride;
    if (mrow[t] == 0) v0 = -3.4028235e38f;
    if (mrow[t + 256] == 0) v1 = -3.4028235e38f;
  }
  __shared__ float red[256];
  red[t] = fmaxf(v0, v1);
  __syncthreads();
  for (int s = 128; s > 0; s >>= 1) {
    if (t < s) red[t] = fmaxf(red[t], red[t + s]);
    __syncthreads();
  }
  const float m = red[0];
  __syncthreads();
  const float e0 = __expf(v0 - m), e1 = __expf(v1 - m);
  red[t] = e0 + e1;
  __syncthreads();
  for (int s = 128; s > 0; s >>= 1) {
    if (t < s) red[t] += red[t + s];
    __syncthreads();
  }
  const float inv = 1.f / red[0];
  bf16_t* dst = P + (long long)row * 512;
  dst[t] = f2bf(e0 * inv);
  dst[t + 256] = f2bf(e1 * inv);
}

// ---------------------------------------------------------------------------
// out = LayerNorm(X + R) * g + b   (row width 512); writes fp32 + bf16
// ---------------------------------------------------------------------------
__global__ void __launch_bounds__(256) add_layernorm(
    const float* __restrict__ X, const float* __restrict__ R,
    const float* __restrict__ g, const float* __restrict__ bta,
    float* __restrict__ outF, bf16_t* __restrict__ outB) {
  const int row = blockIdx.x;
  const int t = threadIdx.x;
  const long long base = (long long)row * 512;
  float x0 = X[base + t] + (R ? R[base + t] : 0.f);
  float x1 = X[base + t + 256] + (R ? R[base + t + 256] : 0.f);
  __shared__ float red[256];
  red[t] = x0 + x1;
  __syncthreads();
  for (int s = 128; s > 0; s >>= 1) {
    if (t < s) red[t] += red[t + s];
    __syncthreads();
  }
  const float mu = red[0] * (1.f / 512.f);
  __syncthreads();
  const float d0 = x0 - mu, d1 = x1 - mu;
  red[t] = d0 * d0 + d1 * d1;
  __syncthreads();
  for (int s = 128; s > 0; s >>= 1) {
    if (t < s) red[t] += red[t + s];
    __syncthreads();
  }
  const float rs = rsqrtf(red[0] * (1.f / 512.f) + 1e-5f);
  const float y0 = d0 * rs * g[t] + bta[t];
  const float y1 = d1 * rs * g[t + 256] + bta[t + 256];
  if (outF) { outF[base + t] = y0; outF[base + t + 256] = y1; }
  if (outB) { outB[base + t] = f2bf(y0); outB[base + t + 256] = f2bf(y1); }
}

// ---------------------------------------------------------------------------
// x = emb[token] + pos_emb[b]  (reference quirk: pos sliced by batch index)
// ---------------------------------------------------------------------------
__global__ void __launch_bounds__(256) embed_pos(
    const int* __restrict__ words, const float* __restrict__ emb,
    const float* __restrict__ pos, float* __restrict__ outF,
    bf16_t* __restrict__ outB, int Srows) {
  const int row = blockIdx.x;
  const int t = threadIdx.x;
  const int b = row / Srows;
  const long long eb = (long long)words[row] * 512;
  const long long pb = (long long)b * 512;
  const long long ob = (long long)row * 512;
  const float v0 = emb[eb + t] + pos[pb + t];
  const float v1 = emb[eb + t + 256] + pos[pb + t + 256];
  outF[ob + t] = v0;
  outF[ob + t + 256] = v1;
  outB[ob + t] = f2bf(v0);
  outB[ob + t + 256] = f2bf(v1);
}

__global__ void __launch_bounds__(256) f32_to_bf16(const float* __restrict__ in,
                                                   bf16_t* __restrict__ out,
                                                   long long n) {
  long long i = (long long)blockIdx.x * blockDim.x + threadIdx.x;
  const long long stride = (long long)gridDim.x * blockDim.x;
  for (; i < n; i += stride) out[i] = f2bf(in[i]);
}

// ---------------------------------------------------------------------------
// Host-side launch helper
// ---------------------------------------------------------------------------
static void run_gemm(bool btrans, const bf16_t* A, const bf16_t* Bm,
                     const float* bias, float* Cf, bf16_t* Cbf, int M, int N,
                     int K, int lda, int ldb, int ldc, int Z, long long aSZ,
                     long long bSZ, long long cSB, long long cSH, int hpz,
                     float alpha, int relu, hipStream_t st) {
  dim3 block(256);
  if (btrans) {
    dim3 grid((N + 127) / 128, (M + 63) / 64, Z);
    gemm_bf16_lds<<<grid, block, 0, st>>>(A, Bm, bias, Cf, Cbf, M, N, K, lda,
                                          ldb, ldc, aSZ, bSZ, cSB, cSH, hpz,
                                          alpha, relu);
  } else {
    dim3 grid((N + 127) / 128, (M + 31) / 32, Z);
    gemm_bf16_bn<<<grid, block, 0, st>>>(A, Bm, bias, Cf, Cbf, M, N, K, lda,
                                         ldb, ldc, aSZ, bSZ, cSB, cSH, hpz,
                                         alpha, relu);
  }
}

extern "C" void kernel_launch(void* const* d_in, const int* in_sizes, int n_in,
                              void* d_out, int out_size, void* d_ws,
                              size_t ws_size, hipStream_t stream) {
  (void)in_sizes; (void)n_in; (void)out_size; (void)ws_size;
  constexpr int Bc = 4, Sc = 512, Tc = 512, Dc = 512, Hc = 8, Lc = 4,
                MIDc = 2048;
  constexpr long long DD = (long long)Dc * Dc;              // 262144
  constexpr long long ROWS = (long long)Bc * Sc;            // 2048
  constexpr long long ACT = ROWS * Dc;                      // 1048576
  constexpr long long SCORES = (long long)Bc * Hc * Sc * Sc;  // 8388608
  constexpr int ZBH = Bc * Hc;                              // 32
  constexpr long long HSTRIDE = (long long)Sc * 64;         // per-head chunk
  constexpr long long SSTRIDE = (long long)Sc * Sc;         // per-(b,h) scores
  constexpr long long BSTRIDE = (long long)Sc * Dc;         // per-batch act

  const int* src_words = (const int*)d_in[0];
  const float* target_words = (const float*)d_in[1];
  const int* src_mask = (const int*)d_in[2];
  const int* target_mask = (const int*)d_in[3];
  const float* emb_w = (const float*)d_in[4];
  const float* pos_emb = (const float*)d_in[5];
  const float* enc_ln_g = (const float*)d_in[6];
  const float* enc_ln_b = (const float*)d_in[7];
  const float* enc_attn_w = (const float*)d_in[8];
  const float* enc_attn_b = (const float*)d_in[9];
  const float* enc_fc1_w = (const float*)d_in[10];
  const float* enc_fc1_b = (const float*)d_in[11];
  const float* enc_fc2_w = (const float*)d_in[12];
  const float* enc_fc2_b = (const float*)d_in[13];
  const float* dec_ln_g = (const float*)d_in[14];
  const float* dec_ln_b = (const float*)d_in[15];
  const float* dec_self_w = (const float*)d_in[16];
  const float* dec_self_b = (const float*)d_in[17];
  const float* dec_src_w = (const float*)d_in[18];
  const float* dec_src_b = (const float*)d_in[19];
  const float* dec_fc1_w = (const float*)d_in[20];
  const float* dec_fc1_b = (const float*)d_in[21];
  const float* dec_fc2_w = (const float*)d_in[22];
  const float* dec_fc2_b = (const float*)d_in[23];
  float* out = (float*)d_out;

  // ---- workspace bump allocator (256B aligned) ----
  char* wsBase = (char*)d_ws;
  size_t off = 0;
  auto alloc = [&](long long elems, int esz) -> char* {
    char* p = wsBase + off;
    off = (off + (size_t)elems * esz + 255) & ~(size_t)255;
    return p;
  };
  bf16_t* wEncAttn = (bf16_t*)alloc((long long)Lc * 4 * DD, 2);
  bf16_t* wEncFc1 = (bf16_t*)alloc((long long)Lc * MIDc * Dc, 2);
  bf16_t* wEncFc2 = (bf16_t*)alloc((long long)Lc * Dc * MIDc, 2);
  bf16_t* wDecSelf = (bf16_t*)alloc((long long)Lc * 4 * DD, 2);
  bf16_t* wDecSrc = (bf16_t*)alloc((long long)Lc * 4 * DD, 2);
  bf16_t* wDecFc1 = (bf16_t*)alloc((long long)Lc * MIDc * Dc, 2);
  bf16_t* wDecFc2 = (bf16_t*)alloc((long long)Lc * Dc * MIDc, 2);
  float* xf = (float*)alloc(ACT, 4);
  bf16_t* xbf = (bf16_t*)alloc(ACT, 2);
  float* yf = (float*)alloc(ACT, 4);
  bf16_t* ybf = (bf16_t*)alloc(ACT, 2);
  float* t1f = (float*)alloc(ACT, 4);
  bf16_t* t1bf = (bf16_t*)alloc(ACT, 2);
  float* t2f = (float*)alloc(ACT, 4);
  bf16_t* t2bf = (bf16_t*)alloc(ACT, 2);
  bf16_t* qbf = (bf16_t*)alloc(ACT, 2);
  bf16_t* kbf = (bf16_t*)alloc(ACT, 2);
  bf16_t* vbf = (bf16_t*)alloc(ACT, 2);
  float* scoresf = (float*)alloc(SCORES, 4);
  bf16_t* probsbf = (bf16_t*)alloc(SCORES, 2);
  bf16_t* ctxbf = (bf16_t*)alloc(ACT, 2);
  float* mhaf = (float*)alloc(ACT, 4);
  bf16_t* hbf = (bf16_t*)alloc(ROWS * MIDc, 2);
  float* ffnf = (float*)alloc(ACT, 4);

  // ---- fp32 -> bf16 weight conversion (each call; deterministic) ----
  auto conv = [&](const float* src, bf16_t* dst, long long n) {
    f32_to_bf16<<<dim3(1024), dim3(256), 0, stream>>>(src, dst, n);
  };
  conv(enc_attn_w, wEncAttn, (long long)Lc * 4 * DD);
  conv(enc_fc1_w, wEncFc1, (long long)Lc * MIDc * Dc);
  conv(enc_fc2_w, wEncFc2, (long long)Lc * Dc * MIDc);
  conv(dec_self_w, wDecSelf, (long long)Lc * 4 * DD);
  conv(dec_src_w, wDecSrc, (long long)Lc * 4 * DD);
  conv(dec_fc1_w, wDecFc1, (long long)Lc * MIDc * Dc);
  conv(dec_fc2_w, wDecFc2, (long long)Lc * Dc * MIDc);
  conv(target_words, ybf, ACT);

  embed_pos<<<dim3((int)ROWS), dim3(256), 0, stream>>>(src_words, emb_w,
                                                       pos_emb, xf, xbf, Sc);

  // ---- multi-head attention block (reference reshape quirk = contiguous
  //      head chunks of the (B,S,D) projection buffers) ----
  auto attention = [&](const bf16_t* qinBf, const bf16_t* kvinBf,
                       const bf16_t* W4, const float* b4, const int* mask,
                       long long mbs, long long mrs, const float* resF,
                       const float* lnG, const float* lnB, float* outF,
                       bf16_t* outBf) {
    // Q, K, V projections (bf16 out only)
    run_gemm(true, qinBf, W4 + 0 * DD, b4 + 0 * Dc, nullptr, qbf, (int)ROWS, Dc,
             Dc, Dc, Dc, Dc, 1, 0, 0, 0, 0, 1, 1.f, 0, stream);
    run_gemm(true, kvinBf, W4 + 1 * DD, b4 + 1 * Dc, nullptr, kbf, (int)ROWS,
             Dc, Dc, Dc, Dc, Dc, 1, 0, 0, 0, 0, 1, 1.f, 0, stream);
    run_gemm(true, kvinBf, W4 + 2 * DD, b4 + 2 * Dc, nullptr, vbf, (int)ROWS,
             Dc, Dc, Dc, Dc, Dc, 1, 0, 0, 0, 0, 1, 1.f, 0, stream);
    // scores[z] = Q_z (512x64) * K_z^T / 8 ; z = b*H + h
    run_gemm(true, qbf, kbf, nullptr, scoresf, nullptr, Sc, Sc, 64, 64, 64, Sc,
             ZBH, HSTRIDE, HSTRIDE, SSTRIDE, 0, 1, 0.125f, 0, stream);
    softmax_mask<<<dim3(ZBH * Sc), dim3(256), 0, stream>>>(
        scoresf, mask, probsbf, Sc, Hc, mbs, mrs);
    // ctx[z] = P_z (512x512) * V_z (512x64); scatter into (B,S,D) at col h*64
    run_gemm(false, probsbf, vbf, nullptr, nullptr, ctxbf, Sc, 64, Sc, Sc, 64,
             Dc, ZBH, SSTRIDE, HSTRIDE, BSTRIDE, 64, Hc, 1.f, 0, stream);
    // output projection (fp32 out)
    run_gemm(true, ctxbf, W4 + 3 * DD, b4 + 3 * Dc, mhaf, nullptr, (int)ROWS,
             Dc, Dc, Dc, Dc, Dc, 1, 0, 0, 0, 0, 1, 1.f, 0, stream);
    add_layernorm<<<dim3((int)ROWS), dim3(256), 0, stream>>>(mhaf, resF, lnG,
                                                             lnB, outF, outBf);
  };

  auto ffn = [&](const bf16_t* inBf, const float* resF, const bf16_t* w1,
                 const float* b1, const bf16_t* w2, const float* b2,
                 const float* lnG, const float* lnB, float* outF,
                 bf16_t* outBf) {
    run_gemm(true, inBf, w1, b1, nullptr, hbf, (int)ROWS, MIDc, Dc, Dc, Dc,
             MIDc, 1, 0, 0, 0, 0, 1, 1.f, 1, stream);
    run_gemm(true, hbf, w2, b2, ffnf, nullptr, (int)ROWS, Dc, MIDc, MIDc, MIDc,
             Dc, 1, 0, 0, 0, 0, 1, 1.f, 0, stream);
    add_layernorm<<<dim3((int)ROWS), dim3(256), 0, stream>>>(ffnf, resF, lnG,
                                                             lnB, outF, outBf);
  };

  // ---- encoder ----
  for (int i = 0; i < Lc; ++i) {
    const bf16_t* W4 = wEncAttn + (long long)i * 4 * DD;
    const float* b4 = enc_attn_b + (long long)i * 4 * Dc;
    const float* g = enc_ln_g + (long long)i * Dc;
    const float* be = enc_ln_b + (long long)i * Dc;
    attention(xbf, xbf, W4, b4, src_mask, (long long)Sc, 0, xf, g, be, t1f,
              t1bf);
    ffn(t1bf, t1f, wEncFc1 + (long long)i * MIDc * Dc,
        enc_fc1_b + (long long)i * MIDc, wEncFc2 + (long long)i * Dc * MIDc,
        enc_fc2_b + (long long)i * Dc, g, be, xf, xbf);
  }

  // ---- decoder (target_words passed raw; first residual uses the input) ----
  const float* yresF = target_words;
  for (int i = 0; i < Lc; ++i) {
    const bf16_t* Wsf = wDecSelf + (long long)i * 4 * DD;
    const float* bsf = dec_self_b + (long long)i * 4 * Dc;
    const bf16_t* Wcr = wDecSrc + (long long)i * 4 * DD;
    const float* bcr = dec_src_b + (long long)i * 4 * Dc;
    const float* g = dec_ln_g + (long long)i * Dc;
    const float* be = dec_ln_b + (long long)i * Dc;
    // masked self-attention
    attention(ybf, ybf, Wsf, bsf, target_mask, (long long)Tc * Tc,
              (long long)Tc, yresF, g, be, t1f, t1bf);
    // cross-attention: Q from decoder state, K/V from encoder output x
    attention(t1bf, xbf, Wcr, bcr, nullptr, 0, 0, t1f, g, be, t2f, t2bf);
    // FFN (+ final layer writes fp32 directly into d_out)
    const bool last = (i == Lc - 1);
    ffn(t2bf, t2f, wDecFc1 + (long long)i * MIDc * Dc,
        dec_fc1_b + (long long)i * MIDc, wDecFc2 + (long long)i * Dc * MIDc,
        dec_fc2_b + (long long)i * Dc, g, be, last ? out : yf, ybf);
    yresF = yf;
  }
}
